// Mamba2Mixer_63900523430449
// MI455X (gfx1250) — matrix operand order repaired and checked by reading, not compiled, not run on hardware
//
#include <hip/hip_runtime.h>

// ---------------- types / helpers ----------------
typedef __attribute__((ext_vector_type(16))) __bf16 v16bf;
typedef __attribute__((ext_vector_type(8)))  float  v8f;
typedef unsigned short u16;

union FragU { uint4 q[2]; v16bf v; };

__device__ __forceinline__ u16 f2bf(float x) {
    unsigned u = __builtin_bit_cast(unsigned, x);
    unsigned r = (u + 0x7fffu + ((u >> 16) & 1u)) >> 16;
    return (u16)r;
}

__device__ __forceinline__ v8f zero8() {
    v8f z = {0.f, 0.f, 0.f, 0.f, 0.f, 0.f, 0.f, 0.f};
    return z;
}

// A operand (16x32 bf16, K-major rows): lane&15 = M row.
// lanes 0-15 hold K {k0..k0+7, k0+16..k0+23}; lanes 16-31 hold K {k0+8..15, k0+24..31}.
__device__ __forceinline__ v16bf load_frag_a(const u16* base, int ld, int row0, int k0) {
    int lane = threadIdx.x & 31;
    const u16* p = base + (size_t)(row0 + (lane & 15)) * (size_t)ld
                        + (size_t)(k0 + ((lane >> 4) << 3));
    FragU u;
    u.q[0] = *(const uint4*)p;
    u.q[1] = *(const uint4*)(p + 16);
    return u.v;
}

// B operand (32x16 bf16): lane&15 = N column; lanes 0-15 hold K {k0..k0+7, k0+16..k0+23},
// lanes 16-31 hold K {k0+8..15, k0+24..31}. Storage must be [N][K] with K contiguous.
__device__ __forceinline__ v16bf load_frag_b(const u16* base, int ld, int row0, int k0) {
    int lane = threadIdx.x & 31;
    const u16* p = base + (size_t)(row0 + (lane & 15)) * (size_t)ld
                        + (size_t)(k0 + ((lane >> 4) << 3));
    FragU u;
    u.q[0] = *(const uint4*)p;
    u.q[1] = *(const uint4*)(p + 16);
    return u.v;
}

__device__ __forceinline__ v8f wmma_bf16(v16bf a, v16bf b, v8f c) {
    return __builtin_amdgcn_wmma_f32_16x16x32_bf16(false, a, false, b, (short)0, c, false, false);
}

// C/D 16x16 f32 layout: element i -> row = row0 + i + (lane>=16 ? 8 : 0), col = col0 + (lane&15)
__device__ __forceinline__ void store_acc(float* C, int ldc, int row0, int col0, v8f a) {
    int lane = threadIdx.x & 31;
    float* p = C + (size_t)(row0 + ((lane >> 4) << 3)) * (size_t)ldc + (col0 + (lane & 15));
#pragma unroll
    for (int i = 0; i < 8; ++i) p[(size_t)i * ldc] = a[i];
}

// ---------------- problem constants ----------------
constexpr int Bsz    = 2;
constexpr int Lseq   = 2048;
constexpr int DM     = 2048;
constexpr int DI     = 4096;     // NHEADS*HEAD_DIM
constexpr int DSTATE = 128;
constexpr int NH     = 64;
constexpr int HD     = 64;
constexpr int CHK    = 256;
constexpr int NC     = Lseq / CHK;          // 8
constexpr int CONVD  = DI + 2 * DSTATE;     // 4352
constexpr int NPROJ  = 2 * DI + 2 * DSTATE + NH; // 8512
constexpr int NPAD   = 8576;                // 67*128, padded N for in_proj GEMM
constexpr int ROWS   = Bsz * Lseq;          // 4096

// ---------------- conversions ----------------
__global__ void f32_to_bf16_kernel(const float* __restrict__ s, u16* __restrict__ d, long long n) {
    long long i = (long long)blockIdx.x * blockDim.x + threadIdx.x;
    if (i < n) d[i] = f2bf(s[i]);
}

__global__ void win_pad_bf16_kernel(const float* __restrict__ s, u16* __restrict__ d) {
    long long i = (long long)blockIdx.x * blockDim.x + threadIdx.x;
    long long n = (long long)NPAD * DM;
    if (i >= n) return;
    long long row = i / DM;
    d[i] = f2bf(row < NPROJ ? s[i] : 0.f);
}

// ---------------- double-buffered LDS-staged NT GEMM: C[M,N] = A[M,K] * B[N,K]^T ----
// BM=128, BN=128, BK=32; 256 threads = 8 waves (wave32); wave tile = 64x32 (8 accums).
__global__ void __launch_bounds__(256) gemm_nt_bf16(
    const u16* __restrict__ A, const u16* __restrict__ B, float* __restrict__ C,
    int M, int N, int K, int lda, int ldb, int ldc,
    long long sA, long long sB, long long sC)
{
    __shared__ u16 As[2][128 * 32];
    __shared__ u16 Bs[2][128 * 32];
    A += (size_t)blockIdx.z * sA;
    B += (size_t)blockIdx.z * sB;
    C += (size_t)blockIdx.z * sC;
    const int m0 = blockIdx.y * 128;
    const int n0 = blockIdx.x * 128;
    const int tid = threadIdx.x;
    const int w  = tid >> 5;
    const int wm = w >> 2;   // 0..1 -> 64 rows
    const int wn = w & 3;    // 0..3 -> 32 cols

    // staging: 128 rows x 32 K per tile; thread covers 32B (16 u16)
    const int r = tid >> 1, half = tid & 1;
    const u16* Ag = A + (size_t)(m0 + r) * (size_t)lda + half * 16;
    const u16* Bg = B + (size_t)(n0 + r) * (size_t)ldb + half * 16;
    const int soff = r * 32 + half * 16;

    uint4 a0, a1, b0, b1;
    a0 = ((const uint4*)Ag)[0]; a1 = ((const uint4*)Ag)[1];
    b0 = ((const uint4*)Bg)[0]; b1 = ((const uint4*)Bg)[1];
    { uint4* s = (uint4*)(&As[0][0] + soff); s[0] = a0; s[1] = a1; }
    { uint4* s = (uint4*)(&Bs[0][0] + soff); s[0] = b0; s[1] = b1; }
    __syncthreads();

    v8f acc[4][2];
#pragma unroll
    for (int i = 0; i < 4; ++i) {
        acc[i][0] = zero8();
        acc[i][1] = zero8();
    }

    const int nk = K >> 5;
    int cur = 0;
    for (int kt = 0; kt < nk; ++kt) {
        if (kt + 1 < nk) {   // issue next tile's global loads before the math
            const u16* Agn = Ag + (size_t)(kt + 1) * 32;
            const u16* Bgn = Bg + (size_t)(kt + 1) * 32;
            a0 = ((const uint4*)Agn)[0]; a1 = ((const uint4*)Agn)[1];
            b0 = ((const uint4*)Bgn)[0]; b1 = ((const uint4*)Bgn)[1];
        }
        const u16* Asr = &As[cur][0];
        const u16* Bsr = &Bs[cur][0];
        v16bf bf0 = load_frag_b(Bsr, 32, wn * 32, 0);
        v16bf bf1 = load_frag_b(Bsr, 32, wn * 32 + 16, 0);
#pragma unroll
        for (int mt = 0; mt < 4; ++mt) {
            v16bf af = load_frag_a(Asr, 32, wm * 64 + mt * 16, 0);
            acc[mt][0] = wmma_bf16(af, bf0, acc[mt][0]);
            acc[mt][1] = wmma_bf16(af, bf1, acc[mt][1]);
        }
        if (kt + 1 < nk) {
            uint4* sa = (uint4*)(&As[cur ^ 1][0] + soff);
            uint4* sb = (uint4*)(&Bs[cur ^ 1][0] + soff);
            sa[0] = a0; sa[1] = a1;
            sb[0] = b0; sb[1] = b1;
            __syncthreads();
            cur ^= 1;
        }
    }
#pragma unroll
    for (int mt = 0; mt < 4; ++mt) {
        store_acc(C, ldc, m0 + wm * 64 + mt * 16, n0 + wn * 32,      acc[mt][0]);
        store_acc(C, ldc, m0 + wm * 64 + mt * 16, n0 + wn * 32 + 16, acc[mt][1]);
    }
}

// ---------------- depthwise causal conv (K=4) + SiLU ----------------
__global__ void conv_silu_kernel(const float* __restrict__ zx,
                                 const float* __restrict__ cw, const float* __restrict__ cb,
                                 float* __restrict__ xconv,
                                 u16* __restrict__ Cmb, u16* __restrict__ Bmb,
                                 u16* __restrict__ BmT)
{
    long long idx = (long long)blockIdx.x * blockDim.x + threadIdx.x;
    if (idx >= (long long)Bsz * Lseq * CONVD) return;
    int ch = (int)(idx % CONVD);
    int l  = (int)((idx / CONVD) % Lseq);
    int b  = (int)(idx / ((long long)CONVD * Lseq));

    float acc = cb[ch];
#pragma unroll
    for (int k = 0; k < 4; ++k) {
        int ls = l - 3 + k;
        if (ls >= 0)
            acc += zx[((size_t)(b * Lseq + ls)) * NPAD + DI + ch] * cw[ch * 4 + k];
    }
    float sv = acc / (1.f + __expf(-acc));
    int row = b * Lseq + l;
    if (ch < DI) {
        xconv[(size_t)row * DI + ch] = sv;
    } else if (ch < DI + DSTATE) {
        int n = ch - DI;
        u16 v = f2bf(sv);
        Bmb[(size_t)row * DSTATE + n] = v;
        int c = l >> 8, t = l & 255;
        BmT[(((size_t)(b * NC + c) * DSTATE + n) * CHK) + t] = v;
    } else {
        Cmb[(size_t)row * DSTATE + (ch - DI - DSTATE)] = f2bf(sv);
    }
}

// ---------------- dt softplus + per-chunk cumsum(dA) ----------------
__global__ void dt_kernel(const float* __restrict__ zx, const float* __restrict__ A,
                          const float* __restrict__ dt_bias,
                          float* __restrict__ dtb, float* __restrict__ Acs,
                          float* __restrict__ cdec)
{
    int idx = blockIdx.x * blockDim.x + threadIdx.x; // (b*8+c)*64 + h
    if (idx >= Bsz * NC * NH) return;
    int h = idx & 63, c = (idx >> 6) & 7, b = idx >> 9;
    float bias = dt_bias[h], a = A[h];
    float cs = 0.f;
    for (int t = 0; t < CHK; ++t) {
        size_t row = (size_t)(b * Lseq + c * CHK + t);
        float x = zx[row * NPAD + 2 * DI + 2 * DSTATE + h] + bias;
        float d = (x > 20.f) ? x : __logf(1.f + __expf(x)); // softplus
        dtb[row * NH + h] = d;
        cs += d * a;
        Acs[(size_t)idx * CHK + t] = cs;
    }
    cdec[idx] = __expf(cs);
}

// ---------------- xdt (and decayed xdt), transposed [b,c,h,p,t] bf16 ----------------
__global__ void xdt_kernel(const float* __restrict__ xconv, const float* __restrict__ dtb,
                           const float* __restrict__ Acs,
                           u16* __restrict__ xdtT, u16* __restrict__ xdtdT)
{
    long long idx = (long long)blockIdx.x * blockDim.x + threadIdx.x;
    if (idx >= (long long)Bsz * NC * NH * HD * CHK) return;
    int t  = (int)(idx & 255);
    int p  = (int)((idx >> 8) & 63);
    int h  = (int)((idx >> 14) & 63);
    int bc = (int)(idx >> 20);      // b*8+c
    int b  = bc >> 3, c = bc & 7;
    size_t row = (size_t)(b * Lseq + c * CHK + t);
    float x = xconv[row * DI + h * HD + p];
    float d = dtb[row * NH + h];
    size_t ab = ((size_t)bc * NH + h) * CHK;
    float dec = __expf(Acs[ab + CHK - 1] - Acs[ab + t]);
    float v = x * d;
    xdtT[idx]  = f2bf(v);
    xdtdT[idx] = f2bf(v * dec);
}

// ---------------- states WMMA: states[b,c,h,p,n] = sum_t xdtd[t,p]*Bm[t,n] ----------------
__global__ void __launch_bounds__(256) states_kernel(
    const u16* __restrict__ xdtdT, const u16* __restrict__ BmT, float* __restrict__ St)
{
    int bch = blockIdx.x;          // (b*8+c)*64 + h
    int bc  = bch >> 6;
    const u16* Ab = xdtdT + (size_t)bch * HD * CHK;     // [p][t]
    const u16* Bb = BmT   + (size_t)bc * DSTATE * CHK;  // [n][t]
    float* Cb = St + (size_t)bch * HD * DSTATE;
    int tid = threadIdx.x, w = tid >> 5;
    int p0 = (w >> 2) * 32, n0 = (w & 3) * 32;
    v8f acc[2][2];
#pragma unroll
    for (int i = 0; i < 2; ++i)
#pragma unroll
        for (int j = 0; j < 2; ++j) acc[i][j] = zero8();
    for (int k0 = 0; k0 < CHK; k0 += 32) {
        v16bf bf0 = load_frag_b(Bb, CHK, n0,      k0);
        v16bf bf1 = load_frag_b(Bb, CHK, n0 + 16, k0);
#pragma unroll
        for (int mt = 0; mt < 2; ++mt) {
            v16bf af = load_frag_a(Ab, CHK, p0 + mt * 16, k0);
            acc[mt][0] = wmma_bf16(af, bf0, acc[mt][0]);
            acc[mt][1] = wmma_bf16(af, bf1, acc[mt][1]);
        }
    }
#pragma unroll
    for (int mt = 0; mt < 2; ++mt)
#pragma unroll
        for (int nt = 0; nt < 2; ++nt)
            store_acc(Cb, DSTATE, p0 + mt * 16, n0 + nt * 16, acc[mt][nt]);
}

// ---------------- sequential chunk scan -> prev states (bf16) ----------------
__global__ void scan_kernel(const float* __restrict__ St, const float* __restrict__ cdec,
                            u16* __restrict__ prevb)
{
    long long idx = (long long)blockIdx.x * blockDim.x + threadIdx.x;
    if (idx >= (long long)Bsz * NH * HD * DSTATE) return;
    int n = (int)(idx & 127);
    int p = (int)((idx >> 7) & 63);
    int h = (int)((idx >> 13) & 63);
    int b = (int)(idx >> 19);
    float S = 0.f;
    for (int c = 0; c < NC; ++c) {
        size_t sidx = ((((size_t)(b * NC + c) * NH + h) * HD + p) * DSTATE + n);
        prevb[sidx] = f2bf(S);
        S = S * cdec[(b * NC + c) * NH + h] + St[sidx];
    }
}

// ---------------- Ydiag: masked-L WMMA, writes Y ----------------
__global__ void __launch_bounds__(256) ydiag_kernel(
    const float* __restrict__ CB, const float* __restrict__ Acs,
    const u16* __restrict__ xdtT, float* __restrict__ Y)
{
    int bch = blockIdx.x;          // (b*8+c)*64 + h
    int h = bch & 63, bc = bch >> 6;
    __shared__ float acs[CHK];
    __shared__ u16 Ms[CHK * 32];
    const float* CBb  = CB + (size_t)bc * CHK * CHK;
    const float* acsg = Acs + (size_t)bch * CHK;
    const u16*   xb   = xdtT + (size_t)bch * HD * CHK; // [p][t]
    float* Yb = Y + (size_t)(bc * CHK) * DI + h * HD;
    int tid = threadIdx.x;
    acs[tid] = acsg[tid];
    __syncthreads();
    int w = tid >> 5;
    int t0 = w * 32;
    v8f acc[2][4];
#pragma unroll
    for (int i = 0; i < 2; ++i)
#pragma unroll
        for (int j = 0; j < 4; ++j) acc[i][j] = zero8();

    for (int s0 = 0; s0 < CHK; s0 += 32) {
        {   // build masked bf16 tile: M[t,s] = (s<=t) ? CB[t,s]*exp(acs[t]-acs[s]) : 0
            int rr = tid;
            float at = acs[rr];
            const float* cbrow = CBb + (size_t)rr * CHK + s0;
            u16* ms = Ms + rr * 32;
#pragma unroll
            for (int j = 0; j < 32; ++j) {
                int s = s0 + j;
                float v = (s <= rr) ? cbrow[j] * __expf(at - acs[s]) : 0.f;
                ms[j] = f2bf(v);
            }
        }
        __syncthreads();
        if (s0 <= t0 + 31) {    // triangular skip
            v16bf bfr[4];
#pragma unroll
            for (int nt = 0; nt < 4; ++nt) bfr[nt] = load_frag_b(xb, CHK, nt * 16, s0);
#pragma unroll
            for (int mt = 0; mt < 2; ++mt) {
                v16bf af = load_frag_a(Ms, 32, t0 + mt * 16, 0);
#pragma unroll
                for (int nt = 0; nt < 4; ++nt)
                    acc[mt][nt] = wmma_bf16(af, bfr[nt], acc[mt][nt]);
            }
        }
        __syncthreads();
    }
#pragma unroll
    for (int mt = 0; mt < 2; ++mt)
#pragma unroll
        for (int nt = 0; nt < 4; ++nt)
            store_acc(Yb, DI, t0 + mt * 16, nt * 16, acc[mt][nt]);
}

// ---------------- Yoff: C @ prev, row-scaled by exp(Acs[t]), accumulated into Y ----------------
__global__ void __launch_bounds__(256) yoff_kernel(
    const u16* __restrict__ Cmb, const u16* __restrict__ prevb,
    const float* __restrict__ Acs, float* __restrict__ Y)
{
    int bch = blockIdx.x;
    int h = bch & 63, bc = bch >> 6;
    const u16* Ab = Cmb   + (size_t)(bc * CHK) * DSTATE;   // [t][n]
    const u16* Bb = prevb + (size_t)bch * HD * DSTATE;     // [p][n]
    const float* acsg = Acs + (size_t)bch * CHK;
    float* Yb = Y + (size_t)(bc * CHK) * DI + h * HD;
    int tid = threadIdx.x, w = tid >> 5;
    int t0 = w * 32;
    v8f acc[2][4];
#pragma unroll
    for (int i = 0; i < 2; ++i)
#pragma unroll
        for (int j = 0; j < 4; ++j) acc[i][j] = zero8();

    for (int k0 = 0; k0 < DSTATE; k0 += 32) {
        v16bf bfr[4];
#pragma unroll
        for (int nt = 0; nt < 4; ++nt) bfr[nt] = load_frag_b(Bb, DSTATE, nt * 16, k0);
#pragma unroll
        for (int mt = 0; mt < 2; ++mt) {
            v16bf af = load_frag_a(Ab, DSTATE, t0 + mt * 16, k0);
#pragma unroll
            for (int nt = 0; nt < 4; ++nt)
                acc[mt][nt] = wmma_bf16(af, bfr[nt], acc[mt][nt]);
        }
    }
    int lane = tid & 31;
#pragma unroll
    for (int mt = 0; mt < 2; ++mt) {
        int rbase = t0 + mt * 16 + ((lane >> 4) << 3);
#pragma unroll
        for (int nt = 0; nt < 4; ++nt) {
            float* p = Yb + (size_t)rbase * DI + nt * 16 + (lane & 15);
#pragma unroll
            for (int i = 0; i < 8; ++i)
                p[(size_t)i * DI] += acc[mt][nt][i] * __expf(acsg[rbase + i]);
        }
    }
}

// ---------------- gated SiLU + RMSNorm -> bf16 ----------------
__global__ void __launch_bounds__(256) gatenorm_kernel(
    const float* __restrict__ Y, const float* __restrict__ xconv,
    const float* __restrict__ zx, const float* __restrict__ Dv,
    const float* __restrict__ norm_w, u16* __restrict__ ynb)
{
    int row = blockIdx.x;
    int tid = threadIdx.x;
    __shared__ float red[256];
    const float* Yr = Y + (size_t)row * DI;
    const float* xr = xconv + (size_t)row * DI;
    const float* zr = zx + (size_t)row * NPAD;
    float yv[16];
    float ss = 0.f;
#pragma unroll
    for (int i = 0; i < 16; ++i) {
        int ch = tid + i * 256;
        float yy = Yr[ch] + xr[ch] * Dv[ch >> 6];
        float z = zr[ch];
        float g = z / (1.f + __expf(-z));
        float v = yy * g;
        yv[i] = v;
        ss += v * v;
    }
    red[tid] = ss;
    __syncthreads();
    for (int s = 128; s > 0; s >>= 1) {
        if (tid < s) red[tid] += red[tid + s];
        __syncthreads();
    }
    float sc = rsqrtf(red[0] / (float)DI + 1e-5f);
    u16* yw = ynb + (size_t)row * DI;
#pragma unroll
    for (int i = 0; i < 16; ++i) {
        int ch = tid + i * 256;
        yw[ch] = f2bf(yv[i] * sc * norm_w[ch]);
    }
}

// ---------------- host launcher ----------------
extern "C" void kernel_launch(void* const* d_in, const int* in_sizes, int n_in,
                              void* d_out, int out_size, void* d_ws, size_t ws_size,
                              hipStream_t stream)
{
    const float* hs      = (const float*)d_in[0];
    const float* W_in    = (const float*)d_in[1];
    const float* conv_w  = (const float*)d_in[2];
    const float* conv_b  = (const float*)d_in[3];
    const float* Av      = (const float*)d_in[4];
    const float* Dv      = (const float*)d_in[5];
    const float* dt_bias = (const float*)d_in[6];
    const float* norm_w  = (const float*)d_in[7];
    const float* W_out   = (const float*)d_in[8];
    float* out = (float*)d_out;

    char* ws = (char*)d_ws;
    size_t off = 0;
    auto alloc = [&](size_t bytes) -> void* {
        void* p = ws + off;
        off += (bytes + 255) & ~(size_t)255;
        return p;
    };

    u16*   hsb    = (u16*)  alloc((size_t)ROWS * DM * 2);
    u16*   Winb   = (u16*)  alloc((size_t)NPAD * DM * 2);
    u16*   Woutb  = (u16*)  alloc((size_t)DM * DI * 2);
    float* zx     = (float*)alloc((size_t)ROWS * NPAD * 4);
    float* xconv  = (float*)alloc((size_t)ROWS * DI * 4);
    u16*   Cmb    = (u16*)  alloc((size_t)ROWS * DSTATE * 2);
    u16*   Bmb    = (u16*)  alloc((size_t)ROWS * DSTATE * 2);
    u16*   BmT    = (u16*)  alloc((size_t)Bsz * NC * DSTATE * CHK * 2);
    float* dtb    = (float*)alloc((size_t)ROWS * NH * 4);
    float* Acs    = (float*)alloc((size_t)Bsz * NC * NH * CHK * 4);
    float* cdec   = (float*)alloc((size_t)Bsz * NC * NH * 4);
    u16*   xdtT   = (u16*)  alloc((size_t)Bsz * NC * NH * HD * CHK * 2);
    u16*   xdtdT  = (u16*)  alloc((size_t)Bsz * NC * NH * HD * CHK * 2);
    float* CBbuf  = (float*)alloc((size_t)Bsz * NC * CHK * CHK * 4);
    float* St     = (float*)alloc((size_t)Bsz * NC * NH * HD * DSTATE * 4);
    u16*   prevb  = (u16*)  alloc((size_t)Bsz * NC * NH * HD * DSTATE * 2);
    float* Ybuf   = (float*)alloc((size_t)ROWS * DI * 4);
    u16*   ynb    = (u16*)  alloc((size_t)ROWS * DI * 2);

    auto nb = [](long long n) { return (unsigned)((n + 255) / 256); };

    // 1) bf16 conversions
    f32_to_bf16_kernel<<<nb((long long)ROWS * DM), 256, 0, stream>>>(hs, hsb, (long long)ROWS * DM);
    win_pad_bf16_kernel<<<nb((long long)NPAD * DM), 256, 0, stream>>>(W_in, Winb);
    f32_to_bf16_kernel<<<nb((long long)DM * DI), 256, 0, stream>>>(W_out, Woutb, (long long)DM * DI);

    // 2) in_proj GEMM: zx[4096, 8576] = hs[4096,2048] * W_in[8576,2048]^T
    gemm_nt_bf16<<<dim3(NPAD / 128, ROWS / 128, 1), 256, 0, stream>>>(
        hsb, Winb, zx, ROWS, NPAD, DM, DM, DM, NPAD, 0, 0, 0);

    // 3) conv + silu
    conv_silu_kernel<<<nb((long long)Bsz * Lseq * CONVD), 256, 0, stream>>>(
        zx, conv_w, conv_b, xconv, Cmb, Bmb, BmT);

    // 4) dt softplus + cumsum
    dt_kernel<<<nb(Bsz * NC * NH), 256, 0, stream>>>(zx, Av, dt_bias, dtb, Acs, cdec);

    // 5) xdt tensors
    xdt_kernel<<<nb((long long)Bsz * NC * NH * HD * CHK), 256, 0, stream>>>(
        xconv, dtb, Acs, xdtT, xdtdT);

    // 6) CB = Cm * Bm^T per (b,c): batched GEMM 256x256x128
    gemm_nt_bf16<<<dim3(CHK / 128, CHK / 128, Bsz * NC), 256, 0, stream>>>(
        Cmb, Bmb, CBbuf, CHK, CHK, DSTATE, DSTATE, DSTATE, CHK,
        (long long)CHK * DSTATE, (long long)CHK * DSTATE, (long long)CHK * CHK);

    // 7) per-chunk states + sequential scan
    states_kernel<<<Bsz * NC * NH, 256, 0, stream>>>(xdtdT, BmT, St);
    scan_kernel<<<nb((long long)Bsz * NH * HD * DSTATE), 256, 0, stream>>>(St, cdec, prevb);

    // 8) Ydiag (writes Y), 9) Yoff (accumulates into Y)
    ydiag_kernel<<<Bsz * NC * NH, 256, 0, stream>>>(CBbuf, Acs, xdtT, Ybuf);
    yoff_kernel<<<Bsz * NC * NH, 256, 0, stream>>>(Cmb, prevb, Acs, Ybuf);

    // 10) gate + RMSNorm -> bf16
    gatenorm_kernel<<<ROWS, 256, 0, stream>>>(Ybuf, xconv, zx, Dv, norm_w, ynb);

    // 11) out_proj GEMM: out[4096,2048] = yn[4096,4096] * W_out[2048,4096]^T
    gemm_nt_bf16<<<dim3(DM / 128, ROWS / 128, 1), 256, 0, stream>>>(
        ynb, Woutb, out, ROWS, DM, DI, DI, DI, DM, 0, 0, 0);

    (void)in_sizes; (void)n_in; (void)out_size; (void)ws_size;
}
